// CNLinkPredictor_51256139711064
// MI455X (gfx1250) — matrix operand order repaired and checked by reading, not compiled, not run on hardware
//
#include <hip/hip_runtime.h>
#include <stdint.h>

#define NNODES 10000
#define DEG    64
#define NEDGE  32768
#define FIN    64
#define HID    256
#define TM     32          // edges per workgroup in fused kernel

typedef unsigned short ushort_t;
typedef __attribute__((ext_vector_type(16))) __bf16 bf16x16;
typedef __attribute__((ext_vector_type(8)))  float  floatx8;

union ABFrag { bf16x16 v; uint4 q[2]; };

// hardware bf16 convert (v_cvt_pk_bf16_f32 class) instead of manual RNE
__device__ __forceinline__ ushort_t f2bf(float f) {
  union { __bf16 b; ushort_t u; } cv;
  cv.b = (__bf16)f;
  return cv.u;
}
__device__ __forceinline__ float bf2f(ushort_t u) {
  union { __bf16 b; ushort_t u; } cv;
  cv.u = u;
  return (float)cv.b;
}

// CDNA5 async global->LDS copy (ASYNCcnt-tracked), 16 bytes per lane
__device__ __forceinline__ void async_copy_b128(const void* gsrc, void* ldst) {
  unsigned      lds = (unsigned)(uintptr_t)ldst;           // low 32b = LDS offset
  unsigned long long ga = (unsigned long long)(uintptr_t)gsrc;
  asm volatile("global_load_async_to_lds_b128 %0, %1, off"
               :: "v"(lds), "v"(ga) : "memory");
}
__device__ __forceinline__ void wait_async0() {
  asm volatile("s_wait_asynccnt 0" ::: "memory");
}

// ---------------------------------------------------------------------------
// Weight prep: fp32 [K][N] -> bf16 [N][K] (so a WMMA B-fragment is contiguous)
// ---------------------------------------------------------------------------
__global__ void wtrans_kernel(const float* __restrict__ w, ushort_t* __restrict__ wt,
                              int K, int N) {
  int t = blockIdx.x * blockDim.x + threadIdx.x;
  if (t >= K * N) return;
  int n = t / K;
  int k = t - n * K;
  wt[t] = f2bf(w[k * N + n]);
}

// ---------------------------------------------------------------------------
// Edge features: wave-per-edge common-neighbor intersection + masked gather.
// Emits bf16 xij[E][64] and xcn[E][64].
// ---------------------------------------------------------------------------
__global__ __launch_bounds__(256) void edge_feat_kernel(
    const float* __restrict__ x, const int* __restrict__ adj,
    const int* __restrict__ tei,
    ushort_t* __restrict__ xij, ushort_t* __restrict__ xcn) {
  int e    = (blockIdx.x * blockDim.x + threadIdx.x) >> 5;   // one wave32 per edge
  int lane = threadIdx.x & 31;
  if (e >= NEDGE) return;

  int i = tei[e];
  int j = tei[NEDGE + e];

  // xij = x[i] * x[j]  (64 features -> 2 per lane)
  float xi0 = x[i * FIN + lane],      xi1 = x[i * FIN + 32 + lane];
  float xj0 = x[j * FIN + lane],      xj1 = x[j * FIN + 32 + lane];
  xij[(size_t)e * FIN + lane]      = f2bf(xi0 * xj0);
  xij[(size_t)e * FIN + 32 + lane] = f2bf(xi1 * xj1);

  // neighbor lists in registers: 2 entries/lane each
  int na0 = adj[i * DEG + lane], na1 = adj[i * DEG + 32 + lane];
  int nb0 = adj[j * DEG + lane], nb1 = adj[j * DEG + 32 + lane];

  // cm[a] = ni[a] in N(j): broadcast each nj via shuffle, packed compare
  int m0 = 0, m1 = 0;
  #pragma unroll 4
  for (int t = 0; t < 32; ++t) {
    int b0 = __shfl(nb0, t, 32);
    int b1 = __shfl(nb1, t, 32);
    m0 |= (na0 == b0) | (na0 == b1);
    m1 |= (na1 == b0) | (na1 == b1);
  }

  // xcn = sum_{a : cm[a]} x[ni[a]]   (mask broadcast -> wave-uniform branch)
  float acc0 = 0.f, acc1 = 0.f;
  for (int a = 0; a < 32; ++a) {
    int nid = __shfl(na0, a, 32);
    int mk  = __shfl(m0,  a, 32);
    if (mk) { acc0 += x[nid * FIN + lane]; acc1 += x[nid * FIN + 32 + lane]; }
    int nid2 = __shfl(na1, a, 32);
    int mk2  = __shfl(m1,  a, 32);
    if (mk2) { acc0 += x[nid2 * FIN + lane]; acc1 += x[nid2 * FIN + 32 + lane]; }
  }
  xcn[(size_t)e * FIN + lane]      = f2bf(acc0);
  xcn[(size_t)e * FIN + 32 + lane] = f2bf(acc1);
}

// ---------------------------------------------------------------------------
// One linear layer on a [TM x K] LDS tile using v_wmma_f32_16x16x32_bf16.
// B strip (weights) register-blocked: loaded once per N-tile, reused for both
// M-tiles. MODE 0: ReLU -> bf16; MODE 1: bf16; MODE 2: d*beta + hij -> bf16.
// ---------------------------------------------------------------------------
template <int MODE, int K>
__device__ __forceinline__ void gemm_layer(
    const ushort_t* sSrc,                   // LDS  [TM][K]  bf16
    const ushort_t* __restrict__ wt,        // glob [256][K] bf16 (transposed)
    const float*    __restrict__ bias,      // glob [256]    f32
    ushort_t* dstB,                         // LDS  [TM][256] bf16
    float betaV, const ushort_t* sHij,      // LDS  [TM][256] bf16 (MODE 2)
    int wv, int lane) {
  const int half = lane >> 4;               // 0: lanes 0-15, 1: lanes 16-31
  const int nl   = lane & 15;
  constexpr int KS = K / 32;
  #pragma unroll
  for (int nsel = 0; nsel < 2; ++nsel) {
    const int nt = wv + nsel * 8;           // this wave's N-tile
    // load B strip once: lanes<16 -> K {0..7,16..23}; lanes>=16 -> K {8..15,24..31}
    const ushort_t* bCol = wt + (size_t)(nt * 16 + nl) * K + half * 8;
    ABFrag bfr[KS];
    #pragma unroll
    for (int ks = 0; ks < KS; ++ks) {
      bfr[ks].q[0] = *(const uint4*)(bCol + ks * 32);
      bfr[ks].q[1] = *(const uint4*)(bCol + ks * 32 + 16);
    }
    #pragma unroll
    for (int mt = 0; mt < 2; ++mt) {
      floatx8 acc = {};
      const int row = mt * 16 + nl;
      // A: lanes<16 -> K {0..7,16..23}; lanes>=16 -> K {8..15,24..31}
      const ushort_t* aRow = sSrc + row * K + half * 8;
      #pragma unroll
      for (int ks = 0; ks < KS; ++ks) {
        ABFrag a;
        a.q[0] = *(const uint4*)(aRow + ks * 32);
        a.q[1] = *(const uint4*)(aRow + ks * 32 + 16);
        acc = __builtin_amdgcn_wmma_f32_16x16x32_bf16(
            false, a.v, false, bfr[ks].v, (short)0, acc, false, false);
      }
      const int   col = nt * 16 + nl;
      const float bv  = bias[col];
      const int   mb  = mt * 16 + half * 8; // D: lane<16 -> M 0..7, else 8..15
      #pragma unroll
      for (int v = 0; v < 8; ++v) {
        float d = acc[v] + bv;
        const int r = mb + v;
        if (MODE == 0) {
          d = fmaxf(d, 0.f);
          dstB[r * HID + col] = f2bf(d);
        } else if (MODE == 1) {
          dstB[r * HID + col] = f2bf(d);
        } else {
          d = d * betaV + bf2f(sHij[r * HID + col]);
          dstB[r * HID + col] = f2bf(d);
        }
      }
    }
  }
}

// ---------------------------------------------------------------------------
// Fused MLP chain: all six linear layers for a 32-edge tile, LDS-resident.
// ---------------------------------------------------------------------------
__global__ __launch_bounds__(256) void fused_mlp_kernel(
    const ushort_t* __restrict__ xij, const ushort_t* __restrict__ xcn,
    const ushort_t* __restrict__ wij1, const float* __restrict__ bij1,
    const ushort_t* __restrict__ wij2, const float* __restrict__ bij2,
    const ushort_t* __restrict__ wcn1, const float* __restrict__ bcn1,
    const ushort_t* __restrict__ wcn2, const float* __restrict__ bcn2,
    const ushort_t* __restrict__ wcn3, const float* __restrict__ bcn3,
    const ushort_t* __restrict__ wlin1, const float* __restrict__ blin1,
    const float* __restrict__ w2, const float* __restrict__ b2,
    const float* __restrict__ betaPtr, float* __restrict__ out) {
  __shared__ ushort_t sA[TM * FIN];         //  4 KB
  __shared__ ushort_t sB0[TM * HID];        // 16 KB
  __shared__ ushort_t sB1[TM * HID];        // 16 KB
  __shared__ ushort_t sHij[TM * HID];       // 16 KB
  const int tid   = threadIdx.x;
  const int wv    = tid >> 5;
  const int lane  = tid & 31;
  const int ebase = blockIdx.x * TM;
  const float betaV = betaPtr[0];

  // ---- xij branch ---- (async global->LDS tile fill: 256 lanes x 16B = 4KB)
  async_copy_b128(xij + (size_t)ebase * FIN + tid * 8, sA + tid * 8);
  wait_async0();
  __syncthreads();
  gemm_layer<0, FIN>(sA, wij1, bij1, sB0, 0.f, nullptr, wv, lane);
  __syncthreads();
  gemm_layer<1, HID>(sB0, wij2, bij2, sHij, 0.f, nullptr, wv, lane);
  __syncthreads();

  // ---- xcn branch ----
  async_copy_b128(xcn + (size_t)ebase * FIN + tid * 8, sA + tid * 8);
  wait_async0();
  __syncthreads();
  gemm_layer<0, FIN>(sA, wcn1, bcn1, sB0, 0.f, nullptr, wv, lane);
  __syncthreads();
  gemm_layer<0, HID>(sB0, wcn2, bcn2, sB1, 0.f, nullptr, wv, lane);
  __syncthreads();
  gemm_layer<2, HID>(sB1, wcn3, bcn3, sB0, betaV, sHij, wv, lane);  // z = hcn*beta + hij
  __syncthreads();

  // ---- head ----
  gemm_layer<0, HID>(sB0, wlin1, blin1, sB1, 0.f, nullptr, wv, lane);
  __syncthreads();

  // 256 -> 1: 8 threads per edge, 32 columns each, shuffle reduce
  const int e = tid >> 3;
  const int s = tid & 7;
  float p = 0.f;
  #pragma unroll 8
  for (int c = s * 32; c < s * 32 + 32; ++c)
    p += bf2f(sB1[e * HID + c]) * w2[c];
  p += __shfl_down(p, 4, 8);
  p += __shfl_down(p, 2, 8);
  p += __shfl_down(p, 1, 8);
  if (s == 0) out[ebase + e] = p + b2[0];
}

// ---------------------------------------------------------------------------
extern "C" void kernel_launch(void* const* d_in, const int* in_sizes, int n_in,
                              void* d_out, int out_size, void* d_ws, size_t ws_size,
                              hipStream_t stream) {
  const float* x       = (const float*)d_in[0];
  const int*   adj     = (const int*)d_in[1];
  const int*   tei     = (const int*)d_in[2];
  const float* beta    = (const float*)d_in[3];
  const float* xcn_w1  = (const float*)d_in[4];
  const float* xcn_b1  = (const float*)d_in[5];
  const float* xcn_w2  = (const float*)d_in[6];
  const float* xcn_b2  = (const float*)d_in[7];
  const float* xcn_w3  = (const float*)d_in[8];
  const float* xcn_b3  = (const float*)d_in[9];
  const float* xij_w1  = (const float*)d_in[10];
  const float* xij_b1  = (const float*)d_in[11];
  const float* xij_w2  = (const float*)d_in[12];
  const float* xij_b2  = (const float*)d_in[13];
  const float* lin_w1  = (const float*)d_in[14];
  const float* lin_b1  = (const float*)d_in[15];
  const float* lin_w2  = (const float*)d_in[16];
  const float* lin_b2  = (const float*)d_in[17];
  float* out = (float*)d_out;

  // workspace layout (bytes)
  char* ws = (char*)d_ws;
  size_t off = 0;
  ushort_t* b_xij  = (ushort_t*)(ws + off); off += (size_t)NEDGE * FIN * 2;
  ushort_t* b_xcn  = (ushort_t*)(ws + off); off += (size_t)NEDGE * FIN * 2;
  ushort_t* t_ij1  = (ushort_t*)(ws + off); off += (size_t)FIN * HID * 2;
  ushort_t* t_ij2  = (ushort_t*)(ws + off); off += (size_t)HID * HID * 2;
  ushort_t* t_cn1  = (ushort_t*)(ws + off); off += (size_t)FIN * HID * 2;
  ushort_t* t_cn2  = (ushort_t*)(ws + off); off += (size_t)HID * HID * 2;
  ushort_t* t_cn3  = (ushort_t*)(ws + off); off += (size_t)HID * HID * 2;
  ushort_t* t_lin1 = (ushort_t*)(ws + off); off += (size_t)HID * HID * 2;
  (void)ws_size; (void)in_sizes; (void)n_in; (void)out_size;

  // 1) weight transpose + bf16 convert
  const int tb = 256;
  wtrans_kernel<<<(FIN * HID + tb - 1) / tb, tb, 0, stream>>>(xij_w1, t_ij1, FIN, HID);
  wtrans_kernel<<<(HID * HID + tb - 1) / tb, tb, 0, stream>>>(xij_w2, t_ij2, HID, HID);
  wtrans_kernel<<<(FIN * HID + tb - 1) / tb, tb, 0, stream>>>(xcn_w1, t_cn1, FIN, HID);
  wtrans_kernel<<<(HID * HID + tb - 1) / tb, tb, 0, stream>>>(xcn_w2, t_cn2, HID, HID);
  wtrans_kernel<<<(HID * HID + tb - 1) / tb, tb, 0, stream>>>(xcn_w3, t_cn3, HID, HID);
  wtrans_kernel<<<(HID * HID + tb - 1) / tb, tb, 0, stream>>>(lin_w1, t_lin1, HID, HID);

  // 2) per-edge intersection + gather (wave32 per edge)
  edge_feat_kernel<<<NEDGE / 8, 256, 0, stream>>>(x, adj, tei, b_xij, b_xcn);

  // 3) fused WMMA MLP chain (32 edges per block)
  fused_mlp_kernel<<<NEDGE / TM, 256, 0, stream>>>(
      b_xij, b_xcn,
      t_ij1, xij_b1, t_ij2, xij_b2,
      t_cn1, xcn_b1, t_cn2, xcn_b2, t_cn3, xcn_b3,
      t_lin1, lin_b1, lin_w2, lin_b2, beta, out);
}
